// lstm_crf_50113678410562
// MI455X (gfx1250) — matrix operand order repaired and checked by reading, not compiled, not run on hardware
//
#include <hip/hip_runtime.h>
#include <hip/hip_bf16.h>

typedef _Float16 h16v __attribute__((ext_vector_type(16)));
typedef _Float16 h8v  __attribute__((ext_vector_type(8)));
typedef float    f8v  __attribute__((ext_vector_type(8)));
typedef int      i32x4 __attribute__((ext_vector_type(4)));

#define BB  32
#define TT  256
#define EE  300
#define KC  64
#define SOSI 2

#if defined(__has_builtin)
#if __has_builtin(__builtin_amdgcn_global_load_async_to_lds_b128) && \
    __has_builtin(__builtin_amdgcn_s_wait_asynccnt)
#define HAVE_ASYNC_LDS 1
#endif
#endif

union AF { h16v v; h8v h[2]; };

__device__ __forceinline__ float sigf(float x) { return 1.f / (1.f + __expf(-x)); }
__device__ __forceinline__ float tanhf_fast(float x) { return 2.f / (1.f + __expf(-2.f * x)) - 1.f; }

// ---------- weight convert fp32 -> f16 with K padding ----------
__global__ void k_convert_pad(const float* __restrict__ src, _Float16* __restrict__ dst,
                              int n, int kin, int kpad) {
    int idx = blockIdx.x * 256 + threadIdx.x;
    if (idx >= n * kpad) return;
    int row = idx / kpad, k = idx - row * kpad;
    float v = (k < kin) ? src[row * kin + k] : 0.f;
    dst[idx] = (_Float16)v;
}

__global__ void k_bias_sum(const float* __restrict__ a, const float* __restrict__ b,
                           float* __restrict__ o, int n) {
    int i = blockIdx.x * 256 + threadIdx.x;
    if (i < n) o[i] = a[i] + b[i];
}

// ---------- embedding + feature concat -> X0 f16 [t*32+b][320] ----------
__global__ void k_embed(const int* __restrict__ x, const float* __restrict__ f,
                        const float* __restrict__ emb, _Float16* __restrict__ X0) {
    int r = blockIdx.x;              // r = t*32 + b
    int t = r >> 5, b = r & 31;
    int tok = x[b * TT + t];
    for (int e = threadIdx.x; e < 320; e += blockDim.x) {
        float v;
        if (e < EE)       v = emb[tok * EE + e];   // embed row 0 is all-zero (PAD)
        else if (e == EE) v = f[b * TT + t];
        else              v = 0.f;
        X0[(size_t)r * 320 + e] = (_Float16)v;
    }
}

// ---------- generic f16 WMMA GEMM: C[M,N] = A[M,Kd] * W[N,Kd]^T + bias (mask opt) ----------
// Async (ASYNCcnt-tracked) global->LDS staging with double-buffered tiles when available.
__global__ __launch_bounds__(256)
void k_gemm(const _Float16* __restrict__ A, int lda,
            const _Float16* __restrict__ W, int ldw,
            const float* __restrict__ bias,
            const int* __restrict__ toks,
            void* __restrict__ Cout, int ldc, int c_f16,
            int M, int N, int Kd) {
#ifdef HAVE_ASYNC_LDS
    __shared__ _Float16 As[2][64 * 32];
    __shared__ _Float16 Bs[2][128 * 32];
#else
    __shared__ _Float16 As[1][64 * 32];
    __shared__ _Float16 Bs[1][128 * 32];
#endif
    int tid = threadIdx.x;
    int lane = tid & 31, wid = tid >> 5;
    int wm = wid & 3, wn = wid >> 2;        // 4x2 wave grid over 64x128 block tile
    int m0 = blockIdx.y * 64;
    int n0 = blockIdx.x * 128;

    f8v acc[4];
    for (int j = 0; j < 4; ++j)
        for (int e = 0; e < 8; ++e) acc[j][e] = 0.f;

#ifdef HAVE_ASYNC_LDS
    // each thread issues exactly 3 async b128 loads per tile (1 A seg + 2 B segs)
    auto stage = [&](int kk, int buf) {
        {
            int row = tid >> 2, part = tid & 3;
            __builtin_amdgcn_global_load_async_to_lds_b128(
                (__attribute__((address_space(1))) i32x4*)(A + (size_t)(m0 + row) * lda + kk + part * 8),
                (__attribute__((address_space(3))) i32x4*)(&As[buf][row * 32 + part * 8]),
                0, 0);
        }
        for (int s = tid; s < 512; s += 256) {
            int row = s >> 2, part = s & 3;
            int n = n0 + row;
            __builtin_amdgcn_global_load_async_to_lds_b128(
                (__attribute__((address_space(1))) i32x4*)(W + (size_t)n * ldw + kk + part * 8),
                (__attribute__((address_space(3))) i32x4*)(&Bs[buf][row * 32 + part * 8]),
                0, 0);
        }
    };
    stage(0, 0);
    int cur = 0;
    for (int kk = 0; kk < Kd; kk += 32) {
        bool more = (kk + 32) < Kd;
        if (more) {
            stage(kk + 32, cur ^ 1);                 // prefetch next tile into alt buffer
            __builtin_amdgcn_s_wait_asynccnt(3);     // current tile's 3 loads complete (in-order)
        } else {
            __builtin_amdgcn_s_wait_asynccnt(0);
        }
        __syncthreads();

        AF a;
        {
            int arow = wm * 16 + (lane & 15);
            int klo = (lane < 16) ? 0 : 8;            // A layout: K 0-7/16-23 | 8-15/24-31
            a.h[0] = *(const h8v*)(&As[cur][arow * 32 + klo]);
            a.h[1] = *(const h8v*)(&As[cur][arow * 32 + 16 + klo]);
        }
        for (int j = 0; j < 4; ++j) {
            AF bf;
            int brow = wn * 64 + j * 16 + (lane & 15);
            int klo = (lane < 16) ? 0 : 8;            // B layout: K 0-7/16-23 | 8-15/24-31
            bf.h[0] = *(const h8v*)(&Bs[cur][brow * 32 + klo]);
            bf.h[1] = *(const h8v*)(&Bs[cur][brow * 32 + 16 + klo]);
            acc[j] = __builtin_amdgcn_wmma_f32_16x16x32_f16(
                false, a.v, false, bf.v, (short)0, acc[j], false, false);
        }
        __syncthreads();
        cur ^= 1;
    }
#else
    for (int kk = 0; kk < Kd; kk += 32) {
        {
            int row = tid >> 2, part = tid & 3;
            h8v va = *(const h8v*)(A + (size_t)(m0 + row) * lda + kk + part * 8);
            *(h8v*)(&As[0][row * 32 + part * 8]) = va;
        }
        for (int s = tid; s < 512; s += 256) {
            int row = s >> 2, part = s & 3;
            int n = n0 + row;
            h8v vb;
            for (int e = 0; e < 8; ++e) vb[e] = (_Float16)0.f;
            if (n < N) vb = *(const h8v*)(W + (size_t)n * ldw + kk + part * 8);
            *(h8v*)(&Bs[0][row * 32 + part * 8]) = vb;
        }
        __syncthreads();
        AF a;
        {
            int arow = wm * 16 + (lane & 15);
            int klo = (lane < 16) ? 0 : 8;
            a.h[0] = *(const h8v*)(&As[0][arow * 32 + klo]);
            a.h[1] = *(const h8v*)(&As[0][arow * 32 + 16 + klo]);
        }
        for (int j = 0; j < 4; ++j) {
            AF bf;
            int brow = wn * 64 + j * 16 + (lane & 15);
            int klo = (lane < 16) ? 0 : 8;
            bf.h[0] = *(const h8v*)(&Bs[0][brow * 32 + klo]);
            bf.h[1] = *(const h8v*)(&Bs[0][brow * 32 + 16 + klo]);
            acc[j] = __builtin_amdgcn_wmma_f32_16x16x32_f16(
                false, a.v, false, bf.v, (short)0, acc[j], false, false);
        }
        __syncthreads();
    }
#endif

    for (int j = 0; j < 4; ++j) {
        int col = n0 + wn * 64 + j * 16 + (lane & 15);
        if (col >= N) continue;
        for (int r = 0; r < 8; ++r) {
            int mrow = m0 + wm * 16 + r + ((lane >> 4) << 3);
            float v = acc[j][r] + bias[col];
            if (toks) {
                int t = mrow >> 5, b = mrow & 31;
                v *= (toks[b * TT + t] > 0) ? 1.f : 0.f;
            }
            if (c_f16) ((_Float16*)Cout)[(size_t)mrow * ldc + col] = (_Float16)v;
            else       ((float*)Cout)[(size_t)mrow * ldc + col] = v;
        }
    }
}

// ---------- persistent bidirectional LSTM step kernel (1 block per direction) ----------
__global__ __launch_bounds__(1024)
void k_lstm(const _Float16* __restrict__ GinF, const _Float16* __restrict__ GinB,
            const _Float16* __restrict__ WhhF, const _Float16* __restrict__ WhhB,
            const int* __restrict__ toks,
            _Float16* __restrict__ Xout) {
    __shared__ _Float16 hS[32 * 256];   // h state (f16, WMMA A operand)
    __shared__ float    cS[32 * 256];   // c state (f32)

    int dir = blockIdx.x;
    const _Float16* Gin = dir ? GinB : GinF;
    const _Float16* Whh = dir ? WhhB : WhhF;
    int outOff = dir ? 256 : 0;

    int tid = threadIdx.x, lane = tid & 31, wid = tid >> 5;
    int mtile = wid & 1;     // batch half (b 0-15 / 16-31)
    int jt = wid >> 1;       // hidden j-tile 0..15; wave owns all 4 gates of its j columns

    for (int i = tid; i < 32 * 256; i += 1024) { hS[i] = (_Float16)0.f; cS[i] = 0.f; }
    __syncthreads();

    for (int step = 0; step < TT; ++step) {
        int t = dir ? (TT - 1 - step) : step;

        // prefetch next timestep's precomputed input-gate slab (L2 hint)
        if (step + 1 < TT) {
            int tn = dir ? (TT - 2 - step) : step + 1;
            __builtin_prefetch(&Gin[((size_t)tn * 32 + (tid & 31)) * 1024], 0, 1);
        }

        // A fragments (h) from LDS, all 8 k-steps up front
        AF af[8];
        {
            int brow = mtile * 16 + (lane & 15);
            int klo = (lane < 16) ? 0 : 8;
            for (int k8 = 0; k8 < 8; ++k8) {
                int kk = k8 * 32;
                af[k8].h[0] = *(const h8v*)(&hS[brow * 256 + kk + klo]);
                af[k8].h[1] = *(const h8v*)(&hS[brow * 256 + kk + 16 + klo]);
            }
        }
        __syncthreads();   // all h reads done before any h writes this step

        f8v acc[4];
        for (int g = 0; g < 4; ++g)
            for (int e = 0; e < 8; ++e) acc[g][e] = 0.f;

        for (int k8 = 0; k8 < 8; ++k8) {
            int kk = k8 * 32;
            for (int g = 0; g < 4; ++g) {            // gates i,f,g,o: rows g*256 + j
                AF bf;
                int n = g * 256 + jt * 16 + (lane & 15);
                const _Float16* wp = Whh + (size_t)n * 256 + kk + ((lane >> 4) << 3);
                bf.h[0] = *(const h8v*)(wp);
                bf.h[1] = *(const h8v*)(wp + 16);
                acc[g] = __builtin_amdgcn_wmma_f32_16x16x32_f16(
                    false, af[k8].v, false, bf.v, (short)0, acc[g], false, false);
            }
        }

        // in-register LSTM cell update: lanes hold matching (b, j) across all 4 gates
        int j = jt * 16 + (lane & 15);
        for (int r = 0; r < 8; ++r) {
            int b = mtile * 16 + r + ((lane >> 4) << 3);
            size_t gbase = ((size_t)t * 32 + b) * 1024;
            float gi = acc[0][r] + (float)Gin[gbase + 0 * 256 + j];
            float gf = acc[1][r] + (float)Gin[gbase + 1 * 256 + j];
            float gg = acc[2][r] + (float)Gin[gbase + 2 * 256 + j];
            float go = acc[3][r] + (float)Gin[gbase + 3 * 256 + j];
            float i_ = sigf(gi), f_ = sigf(gf), g_ = tanhf_fast(gg), o_ = sigf(go);
            float cold = cS[b * 256 + j];
            float hold = (float)hS[b * 256 + j];
            float cnew = f_ * cold + i_ * g_;
            float hnew = o_ * tanhf_fast(cnew);
            float m = (toks[b * TT + t] > 0) ? 1.f : 0.f;
            cS[b * 256 + j] = m * cnew + (1.f - m) * cold;
            hS[b * 256 + j] = (_Float16)(m * hnew + (1.f - m) * hold);
            Xout[((size_t)t * 32 + b) * 512 + outOff + j] = (_Float16)(m * hnew);
        }
        __syncthreads();   // writes visible before next step's h reads
    }
}

// ---------- CRF: Z (forward algorithm) and gold-path score ----------
__global__ __launch_bounds__(64)
void k_crf(const float* __restrict__ Y, const int* __restrict__ toks,
           const int* __restrict__ y0, const float* __restrict__ trans,
           float* __restrict__ out) {
    __shared__ float sc[64];
    __shared__ float red[64];
    int b = blockIdx.x, j = threadIdx.x;

    float trow[64];
    for (int i = 0; i < 64; ++i) trow[i] = trans[j * 64 + i];

    sc[j] = (j == SOSI) ? 0.f : -10000.f;
    __syncthreads();

    for (int t = 0; t < TT; ++t) {
        float m = (toks[b * TT + t] > 0) ? 1.f : 0.f;
        float emit = Y[((size_t)t * 32 + b) * 64 + j];
        float mx = -3.0e38f;
        for (int i = 0; i < 64; ++i) mx = fmaxf(mx, sc[i] + trow[i]);
        float s = 0.f;
        for (int i = 0; i < 64; ++i) s += __expf(sc[i] + trow[i] - mx);
        float nv = emit + mx + __logf(s);
        float res = m * nv + (1.f - m) * sc[j];
        __syncthreads();
        sc[j] = res;
        __syncthreads();
    }

    // Z = logsumexp(sc)
    red[j] = sc[j]; __syncthreads();
    for (int off = 32; off > 0; off >>= 1) { if (j < off) red[j] = fmaxf(red[j], red[j + off]); __syncthreads(); }
    float mz = red[0]; __syncthreads();
    red[j] = __expf(sc[j] - mz); __syncthreads();
    for (int off = 32; off > 0; off >>= 1) { if (j < off) red[j] += red[j + off]; __syncthreads(); }
    float Z = mz + __logf(red[0]);
    __syncthreads();

    // gold path (Y is already mask-multiplied)
    float gp = 0.f;
    for (int t = j; t < TT; t += 64) {
        int cur = y0[b * TT + t];
        int prev = (t == 0) ? SOSI : y0[b * TT + t - 1];
        float m = (toks[b * TT + t] > 0) ? 1.f : 0.f;
        gp += Y[((size_t)t * 32 + b) * 64 + cur] + m * trans[cur * 64 + prev];
    }
    red[j] = gp; __syncthreads();
    for (int off = 32; off > 0; off >>= 1) { if (j < off) red[j] += red[j + off]; __syncthreads(); }
    if (j == 0) out[b] = Z - red[0];
}

extern "C" void kernel_launch(void* const* d_in, const int* in_sizes, int n_in,
                              void* d_out, int out_size, void* d_ws, size_t ws_size,
                              hipStream_t stream) {
    (void)in_sizes; (void)n_in; (void)out_size; (void)ws_size;
    const int*   x     = (const int*)  d_in[0];
    const float* f     = (const float*)d_in[1];
    const int*   y0    = (const int*)  d_in[2];
    const float* emb   = (const float*)d_in[3];
    const float* Wih0f = (const float*)d_in[4];
    const float* Whh0f = (const float*)d_in[5];
    const float* bih0f = (const float*)d_in[6];
    const float* bhh0f = (const float*)d_in[7];
    const float* Wih0b = (const float*)d_in[8];
    const float* Whh0b = (const float*)d_in[9];
    const float* bih0b = (const float*)d_in[10];
    const float* bhh0b = (const float*)d_in[11];
    const float* Wih1f = (const float*)d_in[12];
    const float* Whh1f = (const float*)d_in[13];
    const float* bih1f = (const float*)d_in[14];
    const float* bhh1f = (const float*)d_in[15];
    const float* Wih1b = (const float*)d_in[16];
    const float* Whh1b = (const float*)d_in[17];
    const float* bih1b = (const float*)d_in[18];
    const float* bhh1b = (const float*)d_in[19];
    const float* outw  = (const float*)d_in[20];
    const float* outb  = (const float*)d_in[21];
    const float* trans = (const float*)d_in[22];
    float* out = (float*)d_out;

    char* ws = (char*)d_ws;
    size_t off = 0;
    auto alloc = [&](size_t bytes) -> void* {
        void* p = ws + off;
        off = (off + bytes + 255) & ~(size_t)255;
        return p;
    };
    const size_t R = (size_t)BB * TT;                      // 8192 rows
    _Float16* X0    = (_Float16*)alloc(R * 320 * 2);
    _Float16* X1    = (_Float16*)alloc(R * 512 * 2);
    _Float16* X2    = (_Float16*)alloc(R * 512 * 2);
    _Float16* GinD0 = (_Float16*)alloc(R * 1024 * 2);
    _Float16* GinD1 = (_Float16*)alloc(R * 1024 * 2);
    _Float16* W0f16 = (_Float16*)alloc((size_t)1024 * 320 * 2);
    _Float16* W0b16 = (_Float16*)alloc((size_t)1024 * 320 * 2);
    _Float16* Wh0f  = (_Float16*)alloc((size_t)1024 * 256 * 2);
    _Float16* Wh0b  = (_Float16*)alloc((size_t)1024 * 256 * 2);
    _Float16* Wh1f  = (_Float16*)alloc((size_t)1024 * 256 * 2);
    _Float16* Wh1b  = (_Float16*)alloc((size_t)1024 * 256 * 2);
    _Float16* W1f16 = (_Float16*)alloc((size_t)1024 * 512 * 2);
    _Float16* W1b16 = (_Float16*)alloc((size_t)1024 * 512 * 2);
    _Float16* OutW  = (_Float16*)alloc((size_t)64 * 512 * 2);
    float* bL0f = (float*)alloc(1024 * 4);
    float* bL0b = (float*)alloc(1024 * 4);
    float* bL1f = (float*)alloc(1024 * 4);
    float* bL1b = (float*)alloc(1024 * 4);
    float* Y    = (float*)alloc(R * 64 * 4);

    auto cgrid = [](int total) { return dim3((total + 255) / 256); };

    // weight conversion (fp32 -> f16, K padded to multiple of 32)
    k_convert_pad<<<cgrid(1024 * 320), 256, 0, stream>>>(Wih0f, W0f16, 1024, 301, 320);
    k_convert_pad<<<cgrid(1024 * 320), 256, 0, stream>>>(Wih0b, W0b16, 1024, 301, 320);
    k_convert_pad<<<cgrid(1024 * 256), 256, 0, stream>>>(Whh0f, Wh0f, 1024, 256, 256);
    k_convert_pad<<<cgrid(1024 * 256), 256, 0, stream>>>(Whh0b, Wh0b, 1024, 256, 256);
    k_convert_pad<<<cgrid(1024 * 256), 256, 0, stream>>>(Whh1f, Wh1f, 1024, 256, 256);
    k_convert_pad<<<cgrid(1024 * 256), 256, 0, stream>>>(Whh1b, Wh1b, 1024, 256, 256);
    k_convert_pad<<<cgrid(1024 * 512), 256, 0, stream>>>(Wih1f, W1f16, 1024, 512, 512);
    k_convert_pad<<<cgrid(1024 * 512), 256, 0, stream>>>(Wih1b, W1b16, 1024, 512, 512);
    k_convert_pad<<<cgrid(64 * 512), 256, 0, stream>>>(outw, OutW, 64, 512, 512);
    k_bias_sum<<<cgrid(1024), 256, 0, stream>>>(bih0f, bhh0f, bL0f, 1024);
    k_bias_sum<<<cgrid(1024), 256, 0, stream>>>(bih0b, bhh0b, bL0b, 1024);
    k_bias_sum<<<cgrid(1024), 256, 0, stream>>>(bih1f, bhh1f, bL1f, 1024);
    k_bias_sum<<<cgrid(1024), 256, 0, stream>>>(bih1b, bhh1b, bL1b, 1024);

    // embedding + feature concat
    k_embed<<<R, 128, 0, stream>>>(x, f, emb, X0);

    dim3 gproj((1024 + 127) / 128, (int)(R / 64));   // (8, 128)
    // layer 0 input projections (both directions)
    k_gemm<<<gproj, 256, 0, stream>>>(X0, 320, W0f16, 320, bL0f, nullptr, GinD0, 1024, 1, (int)R, 1024, 320);
    k_gemm<<<gproj, 256, 0, stream>>>(X0, 320, W0b16, 320, bL0b, nullptr, GinD1, 1024, 1, (int)R, 1024, 320);
    // layer 0 recurrence (fwd + bwd in parallel)
    k_lstm<<<2, 1024, 0, stream>>>(GinD0, GinD1, Wh0f, Wh0b, x, X1);
    // layer 1 input projections
    k_gemm<<<gproj, 256, 0, stream>>>(X1, 512, W1f16, 512, bL1f, nullptr, GinD0, 1024, 1, (int)R, 1024, 512);
    k_gemm<<<gproj, 256, 0, stream>>>(X1, 512, W1b16, 512, bL1b, nullptr, GinD1, 1024, 1, (int)R, 1024, 512);
    // layer 1 recurrence
    k_lstm<<<2, 1024, 0, stream>>>(GinD0, GinD1, Wh1f, Wh1b, x, X2);
    // emissions (masked): Y = (X2 @ out_w^T + out_b) * mask
    dim3 gem(1, (int)(R / 64));
    k_gemm<<<gem, 256, 0, stream>>>(X2, 512, OutW, 512, outb, x, Y, 64, 0, (int)R, 64, 512);
    // CRF NLL per sequence
    k_crf<<<BB, 64, 0, stream>>>(Y, x, y0, trans, out);
}
